// SelfAttention_44444321579576
// MI455X (gfx1250) — matrix operand order repaired and checked by reading, not compiled, not run on hardware
//
#include <hip/hip_runtime.h>
#include <hip/hip_bf16.h>
#include <math.h>

typedef _Float16 h16;
typedef __attribute__((ext_vector_type(16))) _Float16 v16h;
typedef __attribute__((ext_vector_type(8)))  float    v8f;

#define Bc   4
#define Sc   2048
#define HIDc 1024
#define NHc  16
#define HDc  64

union AF { v16h v; unsigned u[8]; uint4 q[2]; };

__device__ __forceinline__ v8f wmma_f16(v16h a, v16h b, v8f c) {
  // D = A(16x32 f16) * B(32x16 f16) + C(16x16 f32)
  return __builtin_amdgcn_wmma_f32_16x16x32_f16(false, a, false, b, (short)0, c, false, false);
}

__device__ __forceinline__ v8f zerov8() {
  v8f z = {0.f, 0.f, 0.f, 0.f, 0.f, 0.f, 0.f, 0.f};
  return z;
}

// ---------------------------------------------------------------------------
// Kernel 1: QKV projection.  out[m,n] = sum_k X[m,k] * W[n,k] + bias[n]
// grid = (64, 16, 3)   (128-row blocks of M=8192, 64-col blocks of N=1024, z=q/k/v)
// block = 256 (8 waves); wave w computes rows [16w,16w+16) x 64 cols (4 WMMA accums)
// Outputs f16: q,v -> [b,h,s,d] row-major; k -> kt[b,h,d,s] (transposed for scores)
// ---------------------------------------------------------------------------
__global__ __launch_bounds__(256) void k_qkv(
    const float* __restrict__ X,
    const float* __restrict__ Wq, const float* __restrict__ bq,
    const float* __restrict__ Wk, const float* __restrict__ bk,
    const float* __restrict__ Wv, const float* __restrict__ bv,
    h16* __restrict__ qh, h16* __restrict__ kt, h16* __restrict__ vh)
{
  __shared__ h16 lX[128 * 34];  // [row][k], padded stride 34 (17 dwords, conflict-free)
  __shared__ h16 lW[32 * 66];   // [k][n] transposed, padded stride 66 (33 dwords)

  const int tid  = threadIdx.x;
  const int lane = tid & 31;
  const int w    = tid >> 5;
  const int z    = blockIdx.z;
  const float* Wm = (z == 0) ? Wq : ((z == 1) ? Wk : Wv);
  const float* bm = (z == 0) ? bq : ((z == 1) ? bk : bv);
  const int m0 = blockIdx.x * 128;
  const int n0 = blockIdx.y * 64;

  v8f acc[4];
#pragma unroll
  for (int i = 0; i < 4; ++i) acc[i] = zerov8();

  const int xrow = tid >> 1;          // 0..127
  const int xk   = (tid & 1) * 16;    // 0,16
  const int wrow = tid >> 2;          // 0..63
  const int wk   = (tid & 3) * 8;     // 0,8,16,24

  for (int kc = 0; kc < 32; ++kc) {
    const int k0 = kc * 32;
    // stage X tile (128x32 f32 -> f16)
    const float4* xp = (const float4*)(X + (size_t)(m0 + xrow) * HIDc + k0 + xk);
#pragma unroll
    for (int j = 0; j < 4; ++j) {
      float4 f = xp[j];
      int kk = xk + j * 4;
      lX[xrow * 34 + kk + 0] = (h16)f.x;
      lX[xrow * 34 + kk + 1] = (h16)f.y;
      lX[xrow * 34 + kk + 2] = (h16)f.z;
      lX[xrow * 34 + kk + 3] = (h16)f.w;
    }
    // stage W tile transposed (64x32 f32 -> lW[k][n] f16)
    const float4* wp = (const float4*)(Wm + (size_t)(n0 + wrow) * HIDc + k0 + wk);
#pragma unroll
    for (int j = 0; j < 2; ++j) {
      float4 f = wp[j];
      int kk = wk + j * 4;
      lW[(kk + 0) * 66 + wrow] = (h16)f.x;
      lW[(kk + 1) * 66 + wrow] = (h16)f.y;
      lW[(kk + 2) * 66 + wrow] = (h16)f.z;
      lW[(kk + 3) * 66 + wrow] = (h16)f.w;
    }
    __syncthreads();

    // A fragment: lane -> M = 16w + (lane&15); elem e -> K = e + 8*(e>>3) + 8*(lane>>4)
    AF a;
    const int mrow = 16 * w + (lane & 15);
    const int koff = 8 * (lane >> 4);
#pragma unroll
    for (int v2 = 0; v2 < 8; ++v2) {
      int kk = 2 * v2 + ((v2 >= 4) ? 8 : 0) + koff;
      a.u[v2] = *(const unsigned*)&lX[mrow * 34 + kk];
    }
    // B fragments: lane -> N = nt*16 + (lane&15); elem e -> K = e + 8*(e>>3) + 8*(lane>>4)
    const int kb = (lane & 15) + 16 * (lane >> 4);
#pragma unroll
    for (int nt = 0; nt < 4; ++nt) {
      AF bfr;
#pragma unroll
      for (int i = 0; i < 16; ++i) bfr.v[i] = lW[(i + 8 * (i >> 3) + 8 * (lane >> 4)) * 66 + nt * 16 + (lane & 15)];
      acc[nt] = wmma_f16(a.v, bfr.v, acc[nt]);
    }
    __syncthreads();
  }

  // epilogue: bias add, cvt f16, scatter to head layouts
#pragma unroll
  for (int nt = 0; nt < 4; ++nt) {
    int n_g = n0 + nt * 16 + (lane & 15);   // C/D: lane -> N
    float bia = bm[n_g];
    int hh = n_g >> 6, dd = n_g & 63;
#pragma unroll
    for (int r = 0; r < 8; ++r) {
      int m_g = m0 + 16 * w + r + 8 * (lane >> 4);  // C/D: vgpr r -> M
      int bb = m_g >> 11, ss = m_g & 2047;
      h16 hv = (h16)(acc[nt][r] + bia);
      size_t bh = (size_t)(bb * NHc + hh);
      if (z == 0)      qh[(bh * Sc + ss) * HDc + dd] = hv;
      else if (z == 2) vh[(bh * Sc + ss) * HDc + dd] = hv;
      else             kt[(bh * HDc + dd) * Sc + ss] = hv;
    }
  }
}

// ---------------------------------------------------------------------------
// Kernel 2: fused scores + masks + softmax + probs write + ctx = P @ V
// grid = (128, 16, 4) = (q-tile of 16 rows, head, batch); block = 256 (8 waves)
// Wave w owns key strip [256w, 256w+256): 16 v8f score accumulators.
// Streaming traffic (guide in, probs/ctx out) uses non-temporal cache policy so
// the f16 K/V head slices stay resident in the 192MB L2 across the 128 q-tiles.
// ---------------------------------------------------------------------------
__global__ __launch_bounds__(256) void k_attn(
    const h16* __restrict__ qh, const h16* __restrict__ kt, const h16* __restrict__ vh,
    const float* __restrict__ amask, const float* __restrict__ guide,
    float* __restrict__ ctx, float* __restrict__ probs)
{
  __shared__ h16  lQ[16 * 66];        // Q tile [m][d], padded
  __shared__ h16  lP[8][16 * 34];     // per-wave P chunk (16x32), padded
  __shared__ float red[8][2][8];      // softmax partials [wave][lane-half][r]
  __shared__ float lctx[16 * 64];     // ctx accumulation

  const int tid  = threadIdx.x;
  const int lane = tid & 31;
  const int w    = tid >> 5;
  const int q0   = blockIdx.x * 16;
  const int h    = blockIdx.y;
  const int b    = blockIdx.z;
  const size_t bh = (size_t)b * NHc + h;

  const h16* Qp = qh + (bh * Sc + q0) * HDc;
  const h16* Kp = kt + bh * (size_t)HDc * Sc;
  const h16* Vp = vh + bh * (size_t)Sc * HDc;

  // stage Q tile (1024 halves) + zero lctx
  {
    int i = tid * 4;
    int m = i >> 6, d = i & 63;
    uint2 qv = *(const uint2*)(Qp + i);
    *(unsigned*)&lQ[m * 66 + d]     = qv.x;
    *(unsigned*)&lQ[m * 66 + d + 2] = qv.y;
#pragma unroll
    for (int j = 0; j < 4; ++j) lctx[tid * 4 + j] = 0.f;
  }
  __syncthreads();

  // Q A-fragments for the two 32-wide d-chunks
  AF aq[2];
  {
    int mr = lane & 15, koff = 8 * (lane >> 4);
#pragma unroll
    for (int ch = 0; ch < 2; ++ch)
#pragma unroll
      for (int v2 = 0; v2 < 8; ++v2) {
        int kk = ch * 32 + 2 * v2 + ((v2 >= 4) ? 8 : 0) + koff;
        aq[ch].u[v2] = *(const unsigned*)&lQ[mr * 66 + kk];
      }
  }

  // ---- scores = Q K^T ----
  v8f sc[16];
#pragma unroll
  for (int t = 0; t < 16; ++t) sc[t] = zerov8();

  const int kbl = (lane & 15) + 16 * (lane >> 4);  // B-frag K index within 32
#pragma unroll
  for (int t = 0; t < 16; ++t) {
    int nb = w * 256 + t * 16;
#pragma unroll
    for (int ch = 0; ch < 2; ++ch) {
      AF bf;
#pragma unroll
      for (int i = 0; i < 16; ++i) bf.v[i] = Kp[(size_t)(ch * 32 + i + 8 * (i >> 3) + 8 * (lane >> 4)) * Sc + nb + (lane & 15)];
      sc[t] = wmma_f16(aq[ch].v, bf.v, sc[t]);
    }
  }

  // ---- scale + masks (guide is read-once: non-temporal loads) ----
  const float* gme = guide + (size_t)b * Sc * Sc;
  const float* amr = amask + (size_t)b * Sc;
#pragma unroll
  for (int t = 0; t < 16; ++t) {
    int n = w * 256 + t * 16 + (lane & 15);
    float am = amr[n];
#pragma unroll
    for (int r = 0; r < 8; ++r) {
      int m = q0 + r + 8 * (lane >> 4);
      float g = __builtin_nontemporal_load(&gme[(size_t)m * Sc + n]);
      sc[t][r] = sc[t][r] * 0.125f + am + g;
    }
  }

  // ---- softmax: row max ----
  float rmax[8];
#pragma unroll
  for (int r = 0; r < 8; ++r) {
    float v = -INFINITY;
#pragma unroll
    for (int t = 0; t < 16; ++t) v = fmaxf(v, sc[t][r]);
#pragma unroll
    for (int off = 1; off < 16; off <<= 1) v = fmaxf(v, __shfl_xor(v, off, 32));
    rmax[r] = v;
  }
  if ((lane & 15) == 0) {
#pragma unroll
    for (int r = 0; r < 8; ++r) red[w][lane >> 4][r] = rmax[r];
  }
  __syncthreads();
#pragma unroll
  for (int r = 0; r < 8; ++r) {
    float v = -INFINITY;
    for (int ww = 0; ww < 8; ++ww) v = fmaxf(v, red[ww][lane >> 4][r]);
    rmax[r] = v;
  }
  __syncthreads();

  // ---- softmax: exp + row sum ----
  float rinv[8];
#pragma unroll
  for (int r = 0; r < 8; ++r) {
    float s = 0.f;
#pragma unroll
    for (int t = 0; t < 16; ++t) {
      float e = __expf(sc[t][r] - rmax[r]);
      sc[t][r] = e;
      s += e;
    }
#pragma unroll
    for (int off = 1; off < 16; off <<= 1) s += __shfl_xor(s, off, 32);
    rinv[r] = s;
  }
  if ((lane & 15) == 0) {
#pragma unroll
    for (int r = 0; r < 8; ++r) red[w][lane >> 4][r] = rinv[r];
  }
  __syncthreads();
#pragma unroll
  for (int r = 0; r < 8; ++r) {
    float s = 0.f;
    for (int ww = 0; ww < 8; ++ww) s += red[ww][lane >> 4][r];
    rinv[r] = 1.0f / s;
  }

  // ---- normalize + write probs (write-once output: non-temporal stores) ----
  float* pr = probs + (bh * Sc + (size_t)q0) * Sc;
#pragma unroll
  for (int t = 0; t < 16; ++t) {
    int n = w * 256 + t * 16 + (lane & 15);
#pragma unroll
    for (int r = 0; r < 8; ++r) {
      int m = r + 8 * (lane >> 4);
      float p = sc[t][r] * rinv[r];
      sc[t][r] = p;
      __builtin_nontemporal_store(p, &pr[(size_t)m * Sc + n]);
    }
  }

  // ---- ctx = P @ V (per-wave partial over its 256 keys) ----
  v8f cacc[4];
#pragma unroll
  for (int i = 0; i < 4; ++i) cacc[i] = zerov8();

  h16* myP = &lP[w][0];
#pragma unroll
  for (int c = 0; c < 8; ++c) {
    // dump normalized P chunk (16x32) from score-register layout into LDS
#pragma unroll
    for (int tt = 0; tt < 2; ++tt) {
      int t = 2 * c + tt;
      int nl = tt * 16 + (lane & 15);
#pragma unroll
      for (int r = 0; r < 8; ++r) {
        int m = r + 8 * (lane >> 4);
        myP[m * 34 + nl] = (h16)sc[t][r];
      }
    }
    __asm volatile("s_wait_dscnt 0" ::: "memory");   // wave-private LDS RAW fence
    __builtin_amdgcn_wave_barrier();
    // re-read as A-fragment (transpose via LDS)
    AF ap;
    {
      int mr = lane & 15, koff = 8 * (lane >> 4);
#pragma unroll
      for (int v2 = 0; v2 < 8; ++v2) {
        int kk = 2 * v2 + ((v2 >= 4) ? 8 : 0) + koff;
        ap.u[v2] = *(const unsigned*)&myP[mr * 34 + kk];
      }
    }
    __builtin_amdgcn_wave_barrier();
    // V B-fragments straight from global (row-major [s][d], L2-resident)
    int keyb = w * 256 + c * 32 + kbl;
#pragma unroll
    for (int dt = 0; dt < 4; ++dt) {
      AF bf;
#pragma unroll
      for (int i = 0; i < 16; ++i) bf.v[i] = Vp[(size_t)(w * 256 + c * 32 + i + 8 * (i >> 3) + 8 * (lane >> 4)) * HDc + dt * 16 + (lane & 15)];
      cacc[dt] = wmma_f16(ap.v, bf.v, cacc[dt]);
    }
  }

  // cross-wave reduction of partial ctx via LDS float atomics
#pragma unroll
  for (int dt = 0; dt < 4; ++dt) {
    int d = dt * 16 + (lane & 15);
#pragma unroll
    for (int r = 0; r < 8; ++r) {
      int m = r + 8 * (lane >> 4);
      atomicAdd(&lctx[m * 64 + d], cacc[dt][r]);
    }
  }
  __syncthreads();

  // write ctx: [b, s, h*64 + d]  (write-once output: non-temporal stores)
  {
    float* co = ctx + ((size_t)b * Sc + q0) * HIDc + (size_t)h * HDc;
#pragma unroll
    for (int j = 0; j < 4; ++j) {
      int i = tid * 4 + j;
      int m = i >> 6, d = i & 63;
      __builtin_nontemporal_store(lctx[i], &co[(size_t)m * HIDc + d]);
    }
  }
}

// ---------------------------------------------------------------------------
extern "C" void kernel_launch(void* const* d_in, const int* in_sizes, int n_in,
                              void* d_out, int out_size, void* d_ws, size_t ws_size,
                              hipStream_t stream) {
  const float* X     = (const float*)d_in[0];
  const float* amask = (const float*)d_in[1];
  const float* guide = (const float*)d_in[2];
  const float* Wq    = (const float*)d_in[3];
  const float* bq    = (const float*)d_in[4];
  const float* Wk    = (const float*)d_in[5];
  const float* bk    = (const float*)d_in[6];
  const float* Wv    = (const float*)d_in[7];
  const float* bv    = (const float*)d_in[8];

  float* out   = (float*)d_out;
  float* ctx   = out;                                   // [B,S,HID]
  float* probs = out + (size_t)Bc * Sc * HIDc;          // [B,NH,S,S]

  const size_t headElems = (size_t)Bc * NHc * Sc * HDc; // 8,388,608
  h16* qh = (h16*)d_ws;
  h16* kt = qh + headElems;
  h16* vh = kt + headElems;                             // total 48 MB f16 scratch

  k_qkv<<<dim3(64, 16, 3), 256, 0, stream>>>(X, Wq, bq, Wk, bk, Wv, bv, qh, kt, vh);
  k_attn<<<dim3(128, 16, 4), 256, 0, stream>>>(qh, kt, vh, amask, guide, ctx, probs);
}
